// LinearAttentionLayer_40596030882494
// MI455X (gfx1250) — matrix operand order repaired and checked by reading, not compiled, not run on hardware
//
#include <hip/hip_runtime.h>
#include <hip/hip_bf16.h>

typedef __bf16 bf16_t;
typedef __attribute__((ext_vector_type(16))) __bf16 v16bf;
typedef __attribute__((ext_vector_type(8)))  __bf16 v8bf;
typedef __attribute__((ext_vector_type(8)))  float   v8f;

#define DMODEL 1024
#define NHEADS 16
#define HEADD  64
#define FDIM   64
#define BATCH  4
#define SEQ    4096
#define TOK    (BATCH * SEQ)     // 16384
#define ZBH    (BATCH * NHEADS)  // 64

// ---------- bf16 helpers (RNE convert, bit-level) ----------
__device__ __forceinline__ unsigned short f2bf_u(float f) {
  unsigned int u = __builtin_bit_cast(unsigned int, f);
  u += 0x7fffu + ((u >> 16) & 1u);
  return (unsigned short)(u >> 16);
}
__device__ __forceinline__ bf16_t f2bf(float f) {
  unsigned short s = f2bf_u(f);
  return __builtin_bit_cast(bf16_t, s);
}
__device__ __forceinline__ float bf2f(bf16_t b) {
  unsigned int u = ((unsigned int)__builtin_bit_cast(unsigned short, b)) << 16;
  return __builtin_bit_cast(float, u);
}

// ---------- elementwise f32 -> bf16 ----------
__global__ void convert_f32_bf16(const float* __restrict__ in, bf16_t* __restrict__ out, long n) {
  long i = (long)blockIdx.x * blockDim.x + threadIdx.x;
  if (i < n) out[i] = f2bf(in[i]);
}

// ---------- W [R,C] f32 row-major -> Wt [C,R] bf16 row-major ----------
__global__ void transpose_w(const float* __restrict__ in, bf16_t* __restrict__ out, int R, int C) {
  __shared__ bf16_t tile[32][33];
  int r0 = blockIdx.y * 32, c0 = blockIdx.x * 32;
  #pragma unroll
  for (int i = 0; i < 32; i += 8) {
    int r = r0 + threadIdx.y + i, c = c0 + threadIdx.x;
    tile[threadIdx.y + i][threadIdx.x] = f2bf(in[(long)r * C + c]);
  }
  __syncthreads();
  #pragma unroll
  for (int i = 0; i < 32; i += 8) {
    int c = c0 + threadIdx.y + i, r = r0 + threadIdx.x;
    out[(long)c * R + r] = tile[threadIdx.x][threadIdx.y + i];
  }
}

// ---------- per-head transpose: in[t, h*64+c] -> out[z=(b,h)][c][l]  (bf16) ----------
__global__ void transpose_head(const bf16_t* __restrict__ in, bf16_t* __restrict__ out) {
  __shared__ bf16_t tile[32][33];
  int z = blockIdx.z, b = z / NHEADS, h = z % NHEADS;
  int l0 = blockIdx.x * 32, c0 = blockIdx.y * 32;
  #pragma unroll
  for (int i = 0; i < 32; i += 8) {
    int l = l0 + threadIdx.y + i, c = c0 + threadIdx.x;
    tile[threadIdx.y + i][threadIdx.x] = in[((long)b * SEQ + l) * DMODEL + h * 64 + c];
  }
  __syncthreads();
  #pragma unroll
  for (int i = 0; i < 32; i += 8) {
    int c = c0 + threadIdx.y + i, l = l0 + threadIdx.x;
    out[((long)z * 64 + c) * SEQ + l] = tile[threadIdx.x][threadIdx.y + i];
  }
}

// ---------- ksum[z,f] = sum_l kf[b,l,h*64+f] ----------
__global__ void ksum_kernel(const bf16_t* __restrict__ kf, float* __restrict__ ksum) {
  int z = blockIdx.x, b = z / NHEADS, h = z % NHEADS, f = threadIdx.x;
  const bf16_t* p = kf + (long)b * SEQ * DMODEL + h * 64 + f;
  float s = 0.f;
  for (int l = 0; l < SEQ; ++l) s += bf2f(p[(long)l * DMODEL]);
  ksum[z * 64 + f] = s;
}

// ---------- norm[z,l] = qf[b,l,h,:] . ksum[z,:] ----------
__global__ void norm_kernel(const bf16_t* __restrict__ qf, const float* __restrict__ ksum,
                            float* __restrict__ nrm) {
  int z = blockIdx.y, b = z / NHEADS, h = z % NHEADS;
  int l = blockIdx.x * blockDim.x + threadIdx.x;
  const bf16_t* q = qf + ((long)b * SEQ + l) * DMODEL + h * 64;
  const float* ks = ksum + z * 64;
  float s = 0.f;
  #pragma unroll
  for (int f = 0; f < 64; ++f) s += bf2f(q[f]) * ks[f];
  nrm[(long)z * SEQ + l] = s;
}

// ---------- generic batched bf16 WMMA GEMM: C = A[M,K] * (BT[N,K])^T ----------
// MW = A fragments per wave; wave computes (MW*16) x 64 of C; block = 4 waves,
// block tile = (MW*64) x 64. K-loop is double-buffered (K % 64 == 0) so the
// prefetch for the next 32-wide K-step is in flight while the current WMMAs run.
enum { EP_BF16 = 0, EP_BIAS_RELU_BF16 = 1, EP_DIVNORM_BF16 = 2, EP_BIAS_F32 = 3 };

template <int EP, int MW>
__global__ __launch_bounds__(128) void gemm_bf16_wmma(
    const bf16_t* __restrict__ A, long lda, long aOuter, long aInner,
    const bf16_t* __restrict__ BT, long ldb, long bOuter, long bInner,
    void* __restrict__ Cv, long ldc, long cOuter, long cInner,
    int M, int N, int K, int zdiv,
    const float* __restrict__ bias, const float* __restrict__ nrm, long nrmStride) {
  int z = blockIdx.z;
  long zo = z / zdiv, zi = z % zdiv;
  const bf16_t* Az = A + zo * aOuter + zi * aInner;
  const bf16_t* Bz = BT + zo * bOuter + zi * bInner;

  int lane = threadIdx.x & 31;
  int wave = threadIdx.x >> 5;
  int kh   = lane >> 4;   // half-wave: 0 or 1
  int l15  = lane & 15;
  int mBase = blockIdx.y * (64 * MW) + wave * (16 * MW);
  int nBase = blockIdx.x * 64;

  v8f acc[MW][4] = {};
  // A fragment addressing: lane row = base+l15; contiguous K {0..15} (kh=0) / {16..31} (kh=1)
  const bf16_t* aPtr[MW];
  #pragma unroll
  for (int mi = 0; mi < MW; ++mi)
    aPtr[mi] = Az + (long)(mBase + mi * 16 + l15) * lda + kh * 16;
  // B fragment addressing: lane col = nBase+g*16+l15; contiguous K {0..15} (kh=0) / {16..31} (kh=1)
  const bf16_t* bPtr0 = Bz + (long)(nBase + l15) * ldb + kh * 16;

  auto loadK = [&](int kb, v16bf (&af)[MW], v16bf (&bf4)[4]) {
    #pragma unroll
    for (int mi = 0; mi < MW; ++mi) {
      v8bf a0 = *(const v8bf*)(aPtr[mi] + kb);
      v8bf a1 = *(const v8bf*)(aPtr[mi] + kb + 8);
      #pragma unroll
      for (int i = 0; i < 8; ++i) { af[mi][i] = a0[i]; af[mi][i + 8] = a1[i]; }
    }
    #pragma unroll
    for (int g = 0; g < 4; ++g)
      bf4[g] = *(const v16bf*)(bPtr0 + (long)g * 16 * ldb + kb);
  };
  auto compute = [&](v16bf (&af)[MW], v16bf (&bf4)[4]) {
    #pragma unroll
    for (int g = 0; g < 4; ++g)
      #pragma unroll
      for (int mi = 0; mi < MW; ++mi)
        acc[mi][g] = __builtin_amdgcn_wmma_f32_16x16x32_bf16(false, af[mi], false, bf4[g],
                                                             (short)0, acc[mi][g], false, false);
  };

  v16bf a0[MW], b0[4], a1[MW], b1[4];
  loadK(0, a0, b0);
  for (int kb = 0; kb < K; kb += 64) {       // K % 64 == 0 for all stages
    loadK(kb + 32, a1, b1);                  // prefetch while buffer 0 computes
    compute(a0, b0);
    if (kb + 64 < K) loadK(kb + 64, a0, b0); // prefetch while buffer 1 computes
    compute(a1, b1);
  }

  // Epilogue. C layout: VGPR r -> row base + kh*8 + r, col nBase + g*16 + l15.
  long coff = zo * cOuter + zi * cInner;
  #pragma unroll
  for (int mi = 0; mi < MW; ++mi) {
    int rbase = mBase + mi * 16 + kh * 8;
    float nv[8];
    if constexpr (EP == EP_DIVNORM_BF16) {
      const float* nz = nrm + (long)z * nrmStride;
      #pragma unroll
      for (int r = 0; r < 8; ++r) nv[r] = fmaxf(nz[rbase + r], 1e-6f);
    }
    #pragma unroll
    for (int g = 0; g < 4; ++g) {
      int col = nBase + g * 16 + l15;
      float bv = 0.f;
      if constexpr (EP == EP_BIAS_RELU_BF16 || EP == EP_BIAS_F32) bv = bias[col];
      #pragma unroll
      for (int r = 0; r < 8; ++r) {
        float v = acc[mi][g][r] + bv;
        if constexpr (EP == EP_BIAS_RELU_BF16) v = fmaxf(v, 0.f);
        if constexpr (EP == EP_DIVNORM_BF16) v = v / nv[r];
        long idx = coff + (long)(rbase + r) * ldc + col;
        if constexpr (EP == EP_BIAS_F32) ((float*)Cv)[idx] = v;
        else                             ((bf16_t*)Cv)[idx] = f2bf(v);
      }
    }
  }
}

extern "C" void kernel_launch(void* const* d_in, const int* in_sizes, int n_in,
                              void* d_out, int out_size, void* d_ws, size_t ws_size,
                              hipStream_t stream) {
  (void)in_sizes; (void)n_in; (void)out_size; (void)ws_size;
  const float* x  = (const float*)d_in[0];
  const float* Wq = (const float*)d_in[1];
  const float* Wk = (const float*)d_in[2];
  const float* Wv = (const float*)d_in[3];
  const float* Wo = (const float*)d_in[4];
  const float* bo = (const float*)d_in[5];
  const float* Wf = (const float*)d_in[6];
  const float* bf = (const float*)d_in[7];
  float* out = (float*)d_out;

  char* ws = (char*)d_ws;
  size_t off = 0;
  auto alloc = [&](size_t bytes) { size_t o = off; off += (bytes + 255) & ~(size_t)255; return o; };
  bf16_t* xb  = (bf16_t*)(ws + alloc((size_t)TOK * DMODEL * 2));
  bf16_t* wqT = (bf16_t*)(ws + alloc((size_t)DMODEL * DMODEL * 2));
  bf16_t* wkT = (bf16_t*)(ws + alloc((size_t)DMODEL * DMODEL * 2));
  bf16_t* wvT = (bf16_t*)(ws + alloc((size_t)DMODEL * DMODEL * 2));
  bf16_t* woT = (bf16_t*)(ws + alloc((size_t)DMODEL * DMODEL * 2));
  bf16_t* wfT = (bf16_t*)(ws + alloc((size_t)64 * 64 * 2));
  bf16_t* qb  = (bf16_t*)(ws + alloc((size_t)TOK * DMODEL * 2));
  bf16_t* kb  = (bf16_t*)(ws + alloc((size_t)TOK * DMODEL * 2));
  bf16_t* vb  = (bf16_t*)(ws + alloc((size_t)TOK * DMODEL * 2));
  bf16_t* qfp = (bf16_t*)(ws + alloc((size_t)TOK * DMODEL * 2));
  bf16_t* kfp = (bf16_t*)(ws + alloc((size_t)TOK * DMODEL * 2));
  bf16_t* kfT = (bf16_t*)(ws + alloc((size_t)ZBH * 64 * SEQ * 2));
  bf16_t* vT  = (bf16_t*)(ws + alloc((size_t)ZBH * 64 * SEQ * 2));
  bf16_t* kvT = (bf16_t*)(ws + alloc((size_t)ZBH * 64 * 64 * 2));
  float*  ksm = (float*) (ws + alloc((size_t)ZBH * 64 * 4));
  float*  nrm = (float*) (ws + alloc((size_t)ZBH * SEQ * 4));
  bf16_t* att = (bf16_t*)(ws + alloc((size_t)TOK * DMODEL * 2));

  dim3 tb32(32, 8);
  // 1) convert x and weights to bf16 (weights transposed to [N,K])
  convert_f32_bf16<<<(TOK * DMODEL) / 256, 256, 0, stream>>>(x, xb, (long)TOK * DMODEL);
  transpose_w<<<dim3(32, 32), tb32, 0, stream>>>(Wq, wqT, DMODEL, DMODEL);
  transpose_w<<<dim3(32, 32), tb32, 0, stream>>>(Wk, wkT, DMODEL, DMODEL);
  transpose_w<<<dim3(32, 32), tb32, 0, stream>>>(Wv, wvT, DMODEL, DMODEL);
  transpose_w<<<dim3(32, 32), tb32, 0, stream>>>(Wo, woT, DMODEL, DMODEL);
  transpose_w<<<dim3(2, 2), tb32, 0, stream>>>(Wf, wfT, HEADD, FDIM);

  // 2) Q/K/V projections: [16384,1024] x [1024,1024], 128x64 block tiles
  gemm_bf16_wmma<EP_BF16, 2><<<dim3(DMODEL / 64, TOK / 128, 1), 128, 0, stream>>>(
      xb, DMODEL, 0, 0, wqT, DMODEL, 0, 0, qb, DMODEL, 0, 0, TOK, DMODEL, DMODEL, 1,
      nullptr, nullptr, 0);
  gemm_bf16_wmma<EP_BF16, 2><<<dim3(DMODEL / 64, TOK / 128, 1), 128, 0, stream>>>(
      xb, DMODEL, 0, 0, wkT, DMODEL, 0, 0, kb, DMODEL, 0, 0, TOK, DMODEL, DMODEL, 1,
      nullptr, nullptr, 0);
  gemm_bf16_wmma<EP_BF16, 2><<<dim3(DMODEL / 64, TOK / 128, 1), 128, 0, stream>>>(
      xb, DMODEL, 0, 0, wvT, DMODEL, 0, 0, vb, DMODEL, 0, 0, TOK, DMODEL, DMODEL, 1,
      nullptr, nullptr, 0);

  // 3) feature maps: relu([262144,64] x [64,64] + bf)
  gemm_bf16_wmma<EP_BIAS_RELU_BF16, 2><<<dim3(1, (TOK * NHEADS) / 128, 1), 128, 0, stream>>>(
      qb, 64, 0, 0, wfT, 64, 0, 0, qfp, 64, 0, 0, TOK * NHEADS, FDIM, HEADD, 1,
      bf, nullptr, 0);
  gemm_bf16_wmma<EP_BIAS_RELU_BF16, 2><<<dim3(1, (TOK * NHEADS) / 128, 1), 128, 0, stream>>>(
      kb, 64, 0, 0, wfT, 64, 0, 0, kfp, 64, 0, 0, TOK * NHEADS, FDIM, HEADD, 1,
      bf, nullptr, 0);

  // 4) per-head transposes to [z, c, l] for the K=4096 reduction GEMM
  transpose_head<<<dim3(SEQ / 32, 2, ZBH), tb32, 0, stream>>>(kfp, kfT);
  transpose_head<<<dim3(SEQ / 32, 2, ZBH), tb32, 0, stream>>>(vb, vT);
  ksum_kernel<<<ZBH, 64, 0, stream>>>(kfp, ksm);

  // 5) kvT[z][d,f] = sum_l v[l,d]*kf[l,f]  (A = vT rows d, BT = kfT rows f, K = SEQ)
  gemm_bf16_wmma<EP_BF16, 1><<<dim3(1, 1, ZBH), 128, 0, stream>>>(
      vT, SEQ, (long)64 * SEQ, 0, kfT, SEQ, (long)64 * SEQ, 0, kvT, 64, (long)64 * 64, 0,
      64, 64, SEQ, 1, nullptr, nullptr, 0);

  // 6) normalizer
  norm_kernel<<<dim3(SEQ / 256, ZBH), 256, 0, stream>>>(qfp, ksm, nrm);

  // 7) attn[b,l,h,d] = (qf[b,l,h,:] . kv[b,h,:,d]) / max(norm,1e-6)
  gemm_bf16_wmma<EP_DIVNORM_BF16, 2><<<dim3(1, SEQ / 128, ZBH), 128, 0, stream>>>(
      qfp, DMODEL, (long)SEQ * DMODEL, 64,
      kvT, 64, (long)NHEADS * 64 * 64, (long)64 * 64,
      att, DMODEL, (long)SEQ * DMODEL, 64,
      SEQ, HEADD, FDIM, NHEADS, nullptr, nrm, SEQ);

  // 8) output projection: [16384,1024] x [1024,1024] + bo -> f32
  gemm_bf16_wmma<EP_BIAS_F32, 2><<<dim3(DMODEL / 64, TOK / 128, 1), 128, 0, stream>>>(
      att, DMODEL, 0, 0, woT, DMODEL, 0, 0, out, DMODEL, 0, 0, TOK, DMODEL, DMODEL, 1,
      bo, nullptr, 0);
}
